// Attention_intra_14534169330187
// MI455X (gfx1250) — matrix operand order repaired and checked by reading, not compiled, not run on hardware
//
#include <hip/hip_runtime.h>
#include <hip/hip_bf16.h>
#include <math.h>

// ---------------------------------------------------------------------------
// MI455X / gfx1250 implementation.
// Bandwidth-bound problem (~22 GFLOP vs ~0.6 GB traffic @ 23.3 TB/s), so all
// intermediates are bf16 and every matmul-shaped stage uses
// v_wmma_f32_16x16x32_bf16 (f32 accumulate).
// ---------------------------------------------------------------------------

typedef __attribute__((ext_vector_type(16))) __bf16 v16bf;
typedef __attribute__((ext_vector_type(8)))  float  v8f;

#define HWSZ 65536   // 256*256
#define HDIM 256

// blockified index: n = h1*4+w1 (16 blocks), s = hh*64+ww (4096 per block)
// linear offset inside one (channel) 256x256 plane
__device__ __forceinline__ int blk_addr(int nblk, int s) {
    int h1 = nblk >> 2, w1 = nblk & 3;
    int hh = s >> 6,   ww = s & 63;
    return ((h1 << 6) + hh) * HDIM + (w1 << 6) + ww;
}

// ---------------------------------------------------------------------------
// K1: qkv = conv1x1(x, w_qkv)   (f32 in -> bf16 out)
// GEMM: out[o][p] = sum_i W[o][i] * x[i][p],  o<288, i<96
// One wave = 16 positions; 18 out-channel tiles x 3 K-chunks of WMMA.
// ---------------------------------------------------------------------------
__global__ __launch_bounds__(256) void qkv1x1_kernel(
        const float* __restrict__ x, const float* __restrict__ w,
        __bf16* __restrict__ out) {
    __shared__ __bf16 wl[288 * 96];           // 55 KB bf16 weights
    int tid = threadIdx.x;
    for (int i = tid; i < 288 * 96; i += 256) wl[i] = (__bf16)w[i];
    __syncthreads();

    int b    = blockIdx.y;
    int wave = tid >> 5, lane = tid & 31;
    int hi   = lane >> 4;                     // lane half (K split)
    int arow = lane & 15;
    int p    = blockIdx.x * 128 + wave * 16 + arow;
    const float* xb = x + (size_t)b * 96 * HWSZ;

    // B fragments (x) for the 3 K-chunks, kept live across all out tiles.
    v16bf Bf[3];
#pragma unroll
    for (int kc = 0; kc < 3; ++kc) {
#pragma unroll
        for (int vv = 0; vv < 8; ++vv) {
            int k0 = kc * 32 + hi * 16 + 2 * vv;     // B: K halves split at lane16
            Bf[kc][2 * vv]     = (__bf16)xb[(size_t)k0       * HWSZ + p];
            Bf[kc][2 * vv + 1] = (__bf16)xb[(size_t)(k0 + 1) * HWSZ + p];
        }
    }

    for (int t = 0; t < 18; ++t) {            // 18 tiles of 16 out-channels
        v8f acc = {};
#pragma unroll
        for (int kc = 0; kc < 3; ++kc) {
            v16bf Af;
#pragma unroll
            for (int vv = 0; vv < 8; ++vv) {
                int kk = kc * 32 + hi * 16 + 2 * vv;
                const __bf16* wr = &wl[(t * 16 + arow) * 96 + kk];
                Af[2 * vv] = wr[0]; Af[2 * vv + 1] = wr[1];
            }
            acc = __builtin_amdgcn_wmma_f32_16x16x32_bf16(
                    false, Af, false, Bf[kc], (short)0, acc, false, false);
        }
#pragma unroll
        for (int vv = 0; vv < 8; ++vv) {
            int o = t * 16 + vv + hi * 8;     // D: lane holds N=pos, VGPR holds M=o
            out[((size_t)b * 288 + o) * HWSZ + p] = (__bf16)acc[vv];
        }
    }
}

// ---------------------------------------------------------------------------
// K2: depthwise 3x3, zero pad 1 (bf16 in -> bf16 out). Pure streaming.
// ---------------------------------------------------------------------------
__global__ __launch_bounds__(256) void dw3x3_kernel(
        const __bf16* __restrict__ in, const float* __restrict__ wdw,
        __bf16* __restrict__ out) {
    size_t idx = (size_t)blockIdx.x * 256 + threadIdx.x;   // over 4*288*HW
    int pos = (int)(idx & (HWSZ - 1));
    int c   = (int)((idx >> 16) % 288);
    int r = pos >> 8, col = pos & 255;
    const __bf16* ip = in + (idx - pos);
    const float*  wp = wdw + c * 9;
    float acc = 0.f;
#pragma unroll
    for (int dy = -1; dy <= 1; ++dy)
#pragma unroll
        for (int dx = -1; dx <= 1; ++dx) {
            int rr = r + dy, cc = col + dx;
            if (rr >= 0 && rr < 256 && cc >= 0 && cc < 256)
                acc += wp[(dy + 1) * 3 + (dx + 1)] * (float)ip[rr * HDIM + cc];
        }
    out[idx] = (__bf16)acc;
}

// ---------------------------------------------------------------------------
// K3: per-(b,head,c) channel attention. 1 block = 1 channel, 8 waves.
// Grams via WMMA (unnormalized; l2norm folded in as scalar row/col scaling),
// softmax chain on 16x16 in LDS, then attn@v via WMMA (K padded 16->32).
// ---------------------------------------------------------------------------
__global__ __launch_bounds__(256) void attn_kernel(
        const __bf16* __restrict__ qkvd, const float* __restrict__ mask,
        const float* __restrict__ tx, const float* __restrict__ tm,
        __bf16* __restrict__ outb) {
    __shared__ float nq2[16], nk2[16];
    __shared__ float gx[256], gm[256];
    __shared__ float attnL[256];

    int tid = threadIdx.x, lane = tid & 31, wave = tid >> 5;
    int hi = lane >> 4, arow = lane & 15;
    int cidx = blockIdx.x;                 // 0..383
    int b = cidx / 96, ch = cidx % 96, head = ch / 12;

    const __bf16* q  = qkvd + ((size_t)b * 288 +        ch) * HWSZ;
    const __bf16* k  = qkvd + ((size_t)b * 288 +  96 +  ch) * HWSZ;
    const __bf16* vp = qkvd + ((size_t)b * 288 + 192 +  ch) * HWSZ;
    const float*  mk = mask + ((size_t)b * 96 + ch) * HWSZ;

    if (tid < 16) { nq2[tid] = 0.f; nk2[tid] = 0.f; }
    gx[tid] = 0.f; gm[tid] = 0.f;
    __syncthreads();

    // ---- row norms of q,k (16 threads per row, 256 s each) ----
    {
        int n = tid & 15, js = tid >> 4;
        float sq = 0.f, sk = 0.f;
        for (int j = 0; j < 256; ++j) {
            int a = blk_addr(n, js * 256 + j);
            float qv = (float)q[a], kv = (float)k[a];
            sq += qv * qv; sk += kv * kv;
        }
        atomicAdd(&nq2[n], sq); atomicAdd(&nk2[n], sk);
    }

    // ---- grams: Gx = q.k^T, Gm = m.m^T ; each wave covers 512 of K=4096 ----
    v8f accX = {}, accM = {};
    for (int step = 0; step < 16; ++step) {
        int sbase = wave * 512 + step * 32;   // aligned: s,s+1 contiguous in ww
        v16bf Aq, Am, Bk, Bm;
#pragma unroll
        for (int vv = 0; vv < 8; ++vv) {
            int ka = hi * 16 + 2 * vv;                              // A K map
            int a0 = blk_addr(arow, sbase + ka);
            Aq[2 * vv] = q[a0];            Aq[2 * vv + 1] = q[a0 + 1];
            Am[2 * vv] = (__bf16)mk[a0];   Am[2 * vv + 1] = (__bf16)mk[a0 + 1];
            int kb = hi * 16 + 2 * vv;                              // B K map
            int a1 = blk_addr(arow, sbase + kb);
            Bk[2 * vv] = k[a1];            Bk[2 * vv + 1] = k[a1 + 1];
            Bm[2 * vv] = (__bf16)mk[a1];   Bm[2 * vv + 1] = (__bf16)mk[a1 + 1];
        }
        accX = __builtin_amdgcn_wmma_f32_16x16x32_bf16(
                false, Aq, false, Bk, (short)0, accX, false, false);
        accM = __builtin_amdgcn_wmma_f32_16x16x32_bf16(
                false, Am, false, Bm, (short)0, accM, false, false);
    }
#pragma unroll
    for (int vv = 0; vv < 8; ++vv) {        // reduce 8 wave-partials in LDS
        atomicAdd(&gx[(vv + hi * 8) * 16 + arow], accX[vv]);
        atomicAdd(&gm[(vv + hi * 8) * 16 + arow], accM[vv]);
    }
    __syncthreads();

    // ---- softmax chain; thread owns element (n = tid>>4, m = tid&15) ----
    {
        const float EPS = 1e-12f;
        int n = tid >> 4, m = tid & 15;
        float txh = tx[head], tmh = tm[head];
        float dq = fmaxf(sqrtf(nq2[n]), EPS);
        float dk = fmaxf(sqrtf(nk2[m]), EPS);
        float gxv = gx[tid] / (dq * dk) * txh;  // normalized q.k logit
        float gmv = gm[tid] * tmh;
        gx[tid] = gxv; gm[tid] = gmv;           // own-element overwrite only
        __syncthreads();
        float Mx = -1e30f, Mgm = -1e30f, ssq = 0.f;
        for (int j = 0; j < 16; ++j) {
            float a = gx[n * 16 + j], c = gm[n * 16 + j];
            Mx = fmaxf(Mx, a); Mgm = fmaxf(Mgm, c); ssq += c * c;
        }
        float rn = fmaxf(sqrtf(ssq), EPS);      // _l2norm row scale of attn_m
        float Mm = Mgm / rn;
        float Sx = 0.f, Sm = 0.f;
        for (int j = 0; j < 16; ++j) {
            Sx += __expf(gx[n * 16 + j] - Mx);
            Sm += __expf(gm[n * 16 + j] / rn - Mm);
        }
        float Ma = -1e30f;
        for (int j = 0; j < 16; ++j) {
            float aj = __expf(gx[n * 16 + j] - Mx) / Sx +
                       __expf(gm[n * 16 + j] / rn - Mm) / Sm;
            Ma = fmaxf(Ma, aj);
        }
        float Sa = 0.f;
        for (int j = 0; j < 16; ++j) {
            float aj = __expf(gx[n * 16 + j] - Mx) / Sx +
                       __expf(gm[n * 16 + j] / rn - Mm) / Sm;
            Sa += __expf(aj - Ma);
        }
        float aown = __expf(gxv - Mx) / Sx + __expf(gmv / rn - Mm) / Sm;
        attnL[tid] = __expf(aown - Ma) / Sa;
    }
    __syncthreads();

    // ---- out = attn @ v : WMMA with K padded 16->32 (upper half zero) ----
    v16bf Aa;
#pragma unroll
    for (int vv = 0; vv < 8; ++vv) {
        if (hi == 0) {
            int kk = 2 * vv;                 // real K (m) = 0..15
            Aa[2 * vv]     = (__bf16)attnL[arow * 16 + kk];
            Aa[2 * vv + 1] = (__bf16)attnL[arow * 16 + kk + 1];
        } else {                              // K = 16..31 padding
            Aa[2 * vv] = (__bf16)0.f; Aa[2 * vv + 1] = (__bf16)0.f;
        }
    }
    __bf16* ob = outb + ((size_t)b * 96 + ch) * HWSZ;
    for (int i = 0; i < 32; ++i) {
        int sb = wave * 512 + i * 16;
        v16bf Bv;
#pragma unroll
        for (int vv = 0; vv < 8; ++vv) {
            if (hi == 0) {                    // lanes 16-31 carry K=16..31 -> 0
                int s0 = sb + arow;
                Bv[2 * vv]     = vp[blk_addr(2 * vv,     s0)];
                Bv[2 * vv + 1] = vp[blk_addr(2 * vv + 1, s0)];
            } else {
                Bv[2 * vv] = (__bf16)0.f; Bv[2 * vv + 1] = (__bf16)0.f;
            }
        }
        v8f o = {};
        o = __builtin_amdgcn_wmma_f32_16x16x32_bf16(
                false, Aa, false, Bv, (short)0, o, false, false);
#pragma unroll
        for (int vv = 0; vv < 8; ++vv) {
            int n = vv + hi * 8;
            ob[blk_addr(n, sb + arow)] = (__bf16)o[vv];
        }
    }
}

// ---------------------------------------------------------------------------
// K4: proj = conv1x1 (96x96), bf16 in -> f32 out. Same scheme as K1.
// ---------------------------------------------------------------------------
__global__ __launch_bounds__(256) void proj1x1_kernel(
        const __bf16* __restrict__ in, const float* __restrict__ w,
        float* __restrict__ out) {
    __shared__ __bf16 wl[96 * 96];            // 18 KB
    int tid = threadIdx.x;
    for (int i = tid; i < 96 * 96; i += 256) wl[i] = (__bf16)w[i];
    __syncthreads();

    int b    = blockIdx.y;
    int wave = tid >> 5, lane = tid & 31;
    int hi   = lane >> 4, arow = lane & 15;
    int p    = blockIdx.x * 128 + wave * 16 + arow;
    const __bf16* xb = in + (size_t)b * 96 * HWSZ;

    v16bf Bf[3];
#pragma unroll
    for (int kc = 0; kc < 3; ++kc)
#pragma unroll
        for (int vv = 0; vv < 8; ++vv) {
            int k0 = kc * 32 + hi * 16 + 2 * vv;
            Bf[kc][2 * vv]     = xb[(size_t)k0       * HWSZ + p];
            Bf[kc][2 * vv + 1] = xb[(size_t)(k0 + 1) * HWSZ + p];
        }

    for (int t = 0; t < 6; ++t) {
        v8f acc = {};
#pragma unroll
        for (int kc = 0; kc < 3; ++kc) {
            v16bf Af;
#pragma unroll
            for (int vv = 0; vv < 8; ++vv) {
                int kk = kc * 32 + hi * 16 + 2 * vv;
                const __bf16* wr = &wl[(t * 16 + arow) * 96 + kk];
                Af[2 * vv] = wr[0]; Af[2 * vv + 1] = wr[1];
            }
            acc = __builtin_amdgcn_wmma_f32_16x16x32_bf16(
                    false, Af, false, Bf[kc], (short)0, acc, false, false);
        }
#pragma unroll
        for (int vv = 0; vv < 8; ++vv) {
            int o = t * 16 + vv + hi * 8;
            out[((size_t)b * 96 + o) * HWSZ + p] = acc[vv];
        }
    }
}

// ---------------------------------------------------------------------------
extern "C" void kernel_launch(void* const* d_in, const int* in_sizes, int n_in,
                              void* d_out, int out_size, void* d_ws, size_t ws_size,
                              hipStream_t stream) {
    const float* x      = (const float*)d_in[0];
    const float* mask   = (const float*)d_in[1];
    const float* w_qkv  = (const float*)d_in[2];
    const float* w_dw   = (const float*)d_in[3];
    const float* w_proj = (const float*)d_in[4];
    const float* temp_x = (const float*)d_in[5];
    const float* temp_m = (const float*)d_in[6];

    // workspace: qkv bf16 (151MB) | qkvd bf16 (151MB) | pre-proj out bf16 (50MB)
    __bf16* qkv  = (__bf16*)d_ws;
    __bf16* qkvd = qkv  + (size_t)4 * 288 * HWSZ;
    __bf16* outb = qkvd + (size_t)4 * 288 * HWSZ;
    float*  out  = (float*)d_out;

    dim3 g1(512, 4);   // 512 tiles of 128 positions, 4 batches
    qkv1x1_kernel<<<g1, 256, 0, stream>>>(x, w_qkv, qkv);

    int dwBlocks = (4 * 288 * HWSZ) / 256;   // 294912
    dw3x3_kernel<<<dwBlocks, 256, 0, stream>>>(qkv, w_dw, qkvd);

    attn_kernel<<<384, 256, 0, stream>>>(qkvd, mask, temp_x, temp_m, outb);

    proj1x1_kernel<<<g1, 256, 0, stream>>>(outb, w_proj, out);
}
